// high_Enhancement_48017734369475
// MI455X (gfx1250) — matrix operand order repaired and checked by reading, not compiled, not run on hardware
//
#include <hip/hip_runtime.h>
#include <hip/hip_bf16.h>
#include <math.h>

typedef __attribute__((ext_vector_type(16))) _Float16 v16h;
typedef __attribute__((ext_vector_type(8)))  float    v8f;

#define BB   4
#define CH   64
#define HH   96
#define WW   96
#define HWP  (HH*WW)          // 9216

// ---------------------------------------------------------------------------
// 1) Generic weight swizzle: w (OC, 64, KK) fp32 -> f16 WMMA B-fragments,
//    OC zero-padded to NT*16. K index = kk*64 + c; chunk kt covers 32 K rows.
//    Layout: Bm[((kt*NT + nt)*32 + lane)*16 + e], B-fragment K(e) = 8*half + e + (e>=8 ? 8:0),
//    N = nt*16 + (lane&15).
// ---------------------------------------------------------------------------
__global__ void prep_w_generic(const float* __restrict__ w, _Float16* __restrict__ Bm,
                               int KK, int OC, int NT) {
  int idx = blockIdx.x * blockDim.x + threadIdx.x;
  int total = 2 * KK * NT * 32 * 16;
  if (idx >= total) return;
  int e    = idx & 15;
  int lane = (idx >> 4) & 31;
  int nt   = (idx >> 9) % NT;
  int kt   = (idx >> 9) / NT;
  int o    = nt * 16 + (lane & 15);
  int Kidx = kt * 32 + (lane >> 4) * 8 + e + ((e >= 8) ? 8 : 0);
  int kk   = Kidx / CH;
  int c    = Kidx % CH;
  Bm[idx] = (o < OC) ? (_Float16)w[(size_t)(o * CH + c) * KK + kk] : (_Float16)0.0f;
}

// ---------------------------------------------------------------------------
// 2) Offset convolutions as WMMA GEMM: 3x3 pad 1, 64 -> OC (padded to NT*16).
//    One wave per 16-pixel row tile. The x window (64ch x 3 rows x 24 cols)
//    is staged global->LDS with gfx1250 async-to-LDS copies (ASYNCcnt);
//    out-of-image 16B chunks are zero-filled with ds_store_b128 so the
//    A-fragment reads in the K-loop are UNCONDITIONAL ds_load (no EXEC
//    branching). pos = row + kx + 3 is independent of the tile position.
// ---------------------------------------------------------------------------
template<int NT>
__global__ __launch_bounds__(32)
void offs_wmma_kernel(const float* __restrict__ x,
                      const _Float16* __restrict__ Bmat,
                      const float* __restrict__ bias,
                      float* __restrict__ out, int OC) {
  __shared__ __align__(16) float lds_x[CH * 3 * 24];

  const int lane = threadIdx.x;
  const int tile = blockIdx.x;
  const int p0   = tile * 16;
  const int bb   = p0 / HWP;
  const int pin0 = p0 - bb * HWP;
  const int y    = pin0 / WW;
  const int x0   = pin0 % WW;

  // window covers image columns [csh, csh+24); csh % 4 == 0, may be negative.
  const int csh = x0 - 4;

  // ---- stage x tile: 64ch x 3 rows x 6 chunks of 4 floats (16B) ----
  for (int q = lane; q < CH * 3 * 6; q += 32) {
    int chunk = q % 6;
    int r     = (q / 6) % 3;
    int c     = q / 18;
    int yy    = y - 1 + r;
    int col0  = csh + chunk * 4;
    float* lp = &lds_x[(c * 3 + r) * 24 + chunk * 4];
    // chunks are either fully inside or fully outside the image
    if (yy >= 0 && yy < HH && col0 >= 0 && col0 + 4 <= WW) {
      const float* gp = x + (size_t)(bb * CH + c) * HWP + yy * WW + col0;
      unsigned la = (unsigned)(uintptr_t)lp;
      asm volatile("global_load_async_to_lds_b128 %0, %1, off"
                   :: "v"(la), "v"(gp) : "memory");
    } else {
      *(float4*)lp = make_float4(0.0f, 0.0f, 0.0f, 0.0f);
    }
  }
  asm volatile("s_wait_asynccnt 0" ::: "memory");
  __syncthreads();

  v8f acc[NT] = {};
  const int hlf = lane >> 4;
  const int row = lane & 15;

  for (int kt = 0; kt < 18; ++kt) {
    int kk = kt >> 1;
    int ky = kk / 3, kx = kk % 3;
    int pos = row + kx + 3;          // in [3, 20] always valid
    int c0  = (kt & 1) * 32 + 8 * hlf;

    // A fragment: 16-bit A 16x32 lane layout, K(e) = 8*half + e + (e>=8 ? 8:0)
    v16h a;
#pragma unroll
    for (int e = 0; e < 16; ++e) {
      int c = c0 + e + ((e >= 8) ? 8 : 0);
      a[e] = (_Float16)lds_x[(c * 3 + ky) * 24 + pos];
    }

#pragma unroll
    for (int nt = 0; nt < NT; ++nt) {
      v16h bfr = *(const v16h*)(Bmat + ((size_t)(kt * NT + nt) * 32 + lane) * 16);
      acc[nt] = __builtin_amdgcn_wmma_f32_16x16x32_f16(false, a, false, bfr,
                                                       (short)0, acc[nt], false, false);
    }
  }

  // ---- store (skip padded output channels) ----
#pragma unroll
  for (int nt = 0; nt < NT; ++nt) {
    int oc = nt * 16 + row;
    if (oc < OC) {
      float bv = bias[oc];
      float* ob = out + (size_t)(bb * OC + oc) * HWP + y * WW + x0;
#pragma unroll
      for (int r = 0; r < 8; ++r) ob[hlf * 8 + r] = acc[nt][r] + bv;
    }
  }
}

// ---------------------------------------------------------------------------
// 3) Deformable conv: one wave per 16-pixel tile.
//    Phase A: bilinear corner indices/weights for (16 pixels x KK taps) -> LDS.
//    Phase B: K-loop over 2*KK chunks of 32; gather+blend f16 A-fragment,
//    4x v_wmma_f32_16x16x32_f16 per chunk.
// ---------------------------------------------------------------------------
template<int KS>
__global__ __launch_bounds__(32)
void deform_wmma_kernel(const float* __restrict__ x,
                        const float* __restrict__ off,
                        const _Float16* __restrict__ Bmat,
                        const float* __restrict__ bias,
                        float* __restrict__ out) {
  constexpr int KK  = KS * KS;
  constexpr int PAD = KS / 2;

  __shared__ int4   cidx[KK * 16];
  __shared__ float4 cwt [KK * 16];

  const int lane = threadIdx.x;
  const int tile = blockIdx.x;
  const int p0   = tile * 16;
  const int bb   = p0 / HWP;
  const int pin0 = p0 - bb * HWP;
  const int y    = pin0 / WW;
  const int x0   = pin0 % WW;

  // ---- Phase A: corner metadata ----
  for (int e = lane; e < KK * 16; e += 32) {
    int kk = e >> 4;
    int m  = e & 15;
    int xx = x0 + m;
    int ky = kk / KS, kx = kk % KS;
    const float* offb = off + ((size_t)(bb * 2 * KK) + 2 * kk) * HWP + y * WW + xx;
    float dy = offb[0];
    float dx = offb[HWP];
    float ys = (float)(y + ky - PAD) + dy;
    float xs = (float)(xx + kx - PAD) + dx;
    float y0f = floorf(ys), x0f = floorf(xs);
    float ly = ys - y0f, lx = xs - x0f;
    float hy = 1.0f - ly, hx = 1.0f - lx;
    int iy0 = (int)y0f, ix0 = (int)x0f;
    int iy1 = iy0 + 1,  ix1 = ix0 + 1;
    float vy0 = (iy0 >= 0 && iy0 <= HH - 1) ? 1.0f : 0.0f;
    float vy1 = (iy1 >= 0 && iy1 <= HH - 1) ? 1.0f : 0.0f;
    float vx0 = (ix0 >= 0 && ix0 <= WW - 1) ? 1.0f : 0.0f;
    float vx1 = (ix1 >= 0 && ix1 <= WW - 1) ? 1.0f : 0.0f;
    int cy0 = min(max(iy0, 0), HH - 1), cy1 = min(max(iy1, 0), HH - 1);
    int cx0 = min(max(ix0, 0), WW - 1), cx1 = min(max(ix1, 0), WW - 1);
    cidx[e] = make_int4(cy0 * WW + cx0, cy0 * WW + cx1,
                        cy1 * WW + cx0, cy1 * WW + cx1);
    cwt[e]  = make_float4(hy * hx * vy0 * vx0, hy * lx * vy0 * vx1,
                          ly * hx * vy1 * vx0, ly * lx * vy1 * vx1);
  }
  __syncthreads();

  // ---- Phase B: WMMA K-loop ----
  v8f acc0 = {}, acc1 = {}, acc2 = {}, acc3 = {};
  const int hlf = lane >> 4;
  const int row = lane & 15;
  const float* xb = x + (size_t)(bb * CH) * HWP;

  for (int kt = 0; kt < 2 * KK; ++kt) {
    int kk = kt >> 1;
    int c0 = (kt & 1) * 32 + 8 * hlf;

    int4   ci = cidx[kk * 16 + row];
    float4 cw = cwt [kk * 16 + row];

    if (kt + 1 < 2 * KK)
      __builtin_prefetch(Bmat + ((size_t)((kt + 1) * 4) * 32 + lane) * 16, 0, 1);

    v16h a;
#pragma unroll
    for (int e = 0; e < 16; ++e) {
      int c = c0 + e + ((e >= 8) ? 8 : 0);
      const float* xc = xb + (size_t)c * HWP;
      float v = cw.x * xc[ci.x] + cw.y * xc[ci.y] +
                cw.z * xc[ci.z] + cw.w * xc[ci.w];
      a[e] = (_Float16)v;
    }

    const _Float16* bp = Bmat + ((size_t)(kt * 4) * 32 + lane) * 16;
    v16h b0 = *(const v16h*)(bp);
    v16h b1 = *(const v16h*)(bp + 32 * 16);
    v16h b2 = *(const v16h*)(bp + 2 * 32 * 16);
    v16h b3 = *(const v16h*)(bp + 3 * 32 * 16);

    acc0 = __builtin_amdgcn_wmma_f32_16x16x32_f16(false, a, false, b0, (short)0, acc0, false, false);
    acc1 = __builtin_amdgcn_wmma_f32_16x16x32_f16(false, a, false, b1, (short)0, acc1, false, false);
    acc2 = __builtin_amdgcn_wmma_f32_16x16x32_f16(false, a, false, b2, (short)0, acc2, false, false);
    acc3 = __builtin_amdgcn_wmma_f32_16x16x32_f16(false, a, false, b3, (short)0, acc3, false, false);
  }

  v8f accs[4] = {acc0, acc1, acc2, acc3};
#pragma unroll
  for (int nt = 0; nt < 4; ++nt) {
    int oc = nt * 16 + row;
    float bv = bias[oc];
    float* ob = out + (size_t)(bb * CH + oc) * HWP + y * WW + x0;
#pragma unroll
    for (int r = 0; r < 8; ++r) {
      int m = hlf * 8 + r;
      ob[m] = accs[nt][r] + bv;
    }
  }
}

// ---------------------------------------------------------------------------
// 4) Spatial attention: channel mean/max, then 7x7 conv (2ch) + sigmoid.
// ---------------------------------------------------------------------------
__global__ void spmap_kernel(const float* __restrict__ a1, const float* __restrict__ a2,
                             const float* __restrict__ a3,
                             float* __restrict__ meanm, float* __restrict__ maxm) {
  int idx = blockIdx.x * blockDim.x + threadIdx.x;
  if (idx >= BB * HWP) return;
  int bb = idx / HWP, p = idx % HWP;
  float s = 0.0f, m = -3.4e38f;
  for (int c = 0; c < CH; ++c) {
    size_t o = (size_t)(bb * CH + c) * HWP + p;
    float v1 = a1[o], v2 = a2[o], v3 = a3[o];
    s += v1 + v2 + v3;
    m = fmaxf(m, fmaxf(v1, fmaxf(v2, v3)));
  }
  meanm[idx] = s * (1.0f / 192.0f);
  maxm[idx]  = m;
}

__global__ void spconv_kernel(const float* __restrict__ meanm, const float* __restrict__ maxm,
                              const float* __restrict__ wsp, float* __restrict__ sp) {
  int idx = blockIdx.x * blockDim.x + threadIdx.x;
  if (idx >= BB * HWP) return;
  int bb = idx / HWP, p = idx % HWP;
  int y = p / WW, xx = p % WW;
  float acc = 0.0f;
#pragma unroll
  for (int ky = 0; ky < 7; ++ky) {
    int yy = y + ky - 3;
    if (yy < 0 || yy >= HH) continue;
#pragma unroll
    for (int kx = 0; kx < 7; ++kx) {
      int xc = xx + kx - 3;
      if (xc < 0 || xc >= WW) continue;
      int q = bb * HWP + yy * WW + xc;
      acc += wsp[ky * 7 + kx] * meanm[q] + wsp[49 + ky * 7 + kx] * maxm[q];
    }
  }
  sp[idx] = 1.0f / (1.0f + expf(-acc));
}

// ---------------------------------------------------------------------------
// 5) Channel attention: per-(b, 192ch) mean/max over pixels, then MLP+sigmoid.
// ---------------------------------------------------------------------------
__global__ void chstats_kernel(const float* __restrict__ a1, const float* __restrict__ a2,
                               const float* __restrict__ a3,
                               float* __restrict__ avg, float* __restrict__ mx) {
  int bj = blockIdx.x;
  int bb = bj / 192, j = bj % 192;
  const float* src;
  if (j < 64)        src = a1 + (size_t)(bb * CH + j) * HWP;
  else if (j < 128)  src = a2 + (size_t)(bb * CH + (j - 64)) * HWP;
  else               src = a3 + (size_t)(bb * CH + (j - 128)) * HWP;
  int t = threadIdx.x;
  float s = 0.0f, m = -3.4e38f;
  for (int p = t; p < HWP; p += blockDim.x) {
    float v = src[p];
    s += v;
    m = fmaxf(m, v);
  }
  __shared__ float ss[256], sm[256];
  ss[t] = s; sm[t] = m;
  __syncthreads();
  for (int o = 128; o > 0; o >>= 1) {
    if (t < o) { ss[t] += ss[t + o]; sm[t] = fmaxf(sm[t], sm[t + o]); }
    __syncthreads();
  }
  if (t == 0) { avg[bj] = ss[0] * (1.0f / (float)HWP); mx[bj] = sm[0]; }
}

__global__ void chfc_kernel(const float* __restrict__ avg, const float* __restrict__ mx,
                            const float* __restrict__ w1, const float* __restrict__ w2,
                            float* __restrict__ ch) {
  int bb = blockIdx.x;
  int t = threadIdx.x;
  __shared__ float h1[12];
  if (t < 12) {
    float sa = 0.0f, sm = 0.0f;
    for (int j = 0; j < 192; ++j) {
      float wv = w1[t * 192 + j];
      sa += wv * avg[bb * 192 + j];
      sm += wv * mx[bb * 192 + j];
    }
    h1[t] = fmaxf(sa, 0.0f) + fmaxf(sm, 0.0f);
  }
  __syncthreads();
  if (t < 192) {
    float s = 0.0f;
#pragma unroll
    for (int i = 0; i < 12; ++i) s += w2[t * 12 + i] * h1[i];
    ch[bb * 192 + t] = 1.0f / (1.0f + expf(-s));
  }
}

// ---------------------------------------------------------------------------
// 6) Final fuse: out = a1*(sp+c1) + a2*(sp+c2) + a3*(sp+c3)
// ---------------------------------------------------------------------------
__global__ void final_kernel(const float* __restrict__ a1, const float* __restrict__ a2,
                             const float* __restrict__ a3, const float* __restrict__ sp,
                             const float* __restrict__ ch, float* __restrict__ out) {
  int idx = blockIdx.x * blockDim.x + threadIdx.x;
  if (idx >= BB * CH * HWP) return;
  int p  = idx % HWP;
  int c  = (idx / HWP) % CH;
  int bb = idx / (HWP * CH);
  float spv = sp[bb * HWP + p];
  float c1 = ch[bb * 192 + c];
  float c2 = ch[bb * 192 + 64 + c];
  float c3 = ch[bb * 192 + 128 + c];
  out[idx] = a1[idx] * (spv + c1) + a2[idx] * (spv + c2) + a3[idx] * (spv + c3);
}

// ---------------------------------------------------------------------------
extern "C" void kernel_launch(void* const* d_in, const int* in_sizes, int n_in,
                              void* d_out, int out_size, void* d_ws, size_t ws_size,
                              hipStream_t stream) {
  const float* x      = (const float*)d_in[0];
  const float* w_off3 = (const float*)d_in[1];
  const float* b_off3 = (const float*)d_in[2];
  const float* w_off5 = (const float*)d_in[3];
  const float* b_off5 = (const float*)d_in[4];
  const float* w_off7 = (const float*)d_in[5];
  const float* b_off7 = (const float*)d_in[6];
  const float* w_d3   = (const float*)d_in[7];
  const float* b_d3   = (const float*)d_in[8];
  const float* w_d5   = (const float*)d_in[9];
  const float* b_d5   = (const float*)d_in[10];
  const float* w_d7   = (const float*)d_in[11];
  const float* b_d7   = (const float*)d_in[12];
  const float* w_sp   = (const float*)d_in[13];
  const float* w_ca1  = (const float*)d_in[14];
  const float* w_ca2  = (const float*)d_in[15];
  float* out = (float*)d_out;

  char* ws = (char*)d_ws;
  size_t cur = 0;
  auto alloc = [&](size_t bytes) -> char* {
    cur = (cur + 255) & ~(size_t)255;
    char* p = ws + cur;
    cur += bytes;
    return p;
  };

  float* off3 = (float*)alloc((size_t)BB * 18 * HWP * sizeof(float));
  float* off5 = (float*)alloc((size_t)BB * 50 * HWP * sizeof(float));
  float* off7 = (float*)alloc((size_t)BB * 98 * HWP * sizeof(float));
  float* a1   = (float*)alloc((size_t)BB * CH * HWP * sizeof(float));
  float* a2   = (float*)alloc((size_t)BB * CH * HWP * sizeof(float));
  float* a3   = (float*)alloc((size_t)BB * CH * HWP * sizeof(float));
  _Float16* Bm3 = (_Float16*)alloc((size_t)2 * 9  * 4 * 512 * sizeof(_Float16));
  _Float16* Bm5 = (_Float16*)alloc((size_t)2 * 25 * 4 * 512 * sizeof(_Float16));
  _Float16* Bm7 = (_Float16*)alloc((size_t)2 * 49 * 4 * 512 * sizeof(_Float16));
  _Float16* Bo3 = (_Float16*)alloc((size_t)2 * 9  * 2 * 512 * sizeof(_Float16));
  _Float16* Bo5 = (_Float16*)alloc((size_t)2 * 9  * 4 * 512 * sizeof(_Float16));
  _Float16* Bo7 = (_Float16*)alloc((size_t)2 * 9  * 7 * 512 * sizeof(_Float16));
  float* meanm = (float*)alloc((size_t)BB * HWP * sizeof(float));
  float* maxm  = (float*)alloc((size_t)BB * HWP * sizeof(float));
  float* sp    = (float*)alloc((size_t)BB * HWP * sizeof(float));
  float* avg   = (float*)alloc((size_t)BB * 192 * sizeof(float));
  float* mx    = (float*)alloc((size_t)BB * 192 * sizeof(float));
  float* chb   = (float*)alloc((size_t)BB * 192 * sizeof(float));

  // 1) weight swizzles (deform + offset convs)
  {
    auto launch_prep = [&](const float* w, _Float16* bm, int KK, int OC, int NT) {
      int total = 2 * KK * NT * 512;
      prep_w_generic<<<(total + 255) / 256, 256, 0, stream>>>(w, bm, KK, OC, NT);
    };
    launch_prep(w_d3,   Bm3, 9,  64, 4);
    launch_prep(w_d5,   Bm5, 25, 64, 4);
    launch_prep(w_d7,   Bm7, 49, 64, 4);
    launch_prep(w_off3, Bo3, 9,  18, 2);
    launch_prep(w_off5, Bo5, 9,  50, 4);
    launch_prep(w_off7, Bo7, 9,  98, 7);
  }

  int tiles = BB * HWP / 16;   // 2304

  // 2) offset convs via WMMA (async-to-LDS staged x tiles)
  offs_wmma_kernel<2><<<tiles, 32, 0, stream>>>(x, Bo3, b_off3, off3, 18);
  offs_wmma_kernel<4><<<tiles, 32, 0, stream>>>(x, Bo5, b_off5, off5, 50);
  offs_wmma_kernel<7><<<tiles, 32, 0, stream>>>(x, Bo7, b_off7, off7, 98);

  // 3) deformable convs via WMMA
  deform_wmma_kernel<3><<<tiles, 32, 0, stream>>>(x, off3, Bm3, b_d3, a1);
  deform_wmma_kernel<5><<<tiles, 32, 0, stream>>>(x, off5, Bm5, b_d5, a2);
  deform_wmma_kernel<7><<<tiles, 32, 0, stream>>>(x, off7, Bm7, b_d7, a3);

  // 4) spatial attention
  {
    int t = BB * HWP;
    spmap_kernel <<<(t + 255) / 256, 256, 0, stream>>>(a1, a2, a3, meanm, maxm);
    spconv_kernel<<<(t + 255) / 256, 256, 0, stream>>>(meanm, maxm, w_sp, sp);
  }

  // 5) channel attention
  chstats_kernel<<<BB * 192, 256, 0, stream>>>(a1, a2, a3, avg, mx);
  chfc_kernel<<<BB, 192, 0, stream>>>(avg, mx, w_ca1, w_ca2, chb);

  // 6) fuse
  {
    int t = BB * CH * HWP;
    final_kernel<<<(t + 255) / 256, 256, 0, stream>>>(a1, a2, a3, sp, chb, out);
  }
}
